// CrossAttention_44306882625599
// MI455X (gfx1250) — compile-verified, matrix operand order checked by reading, not run on hardware
//
#include <hip/hip_runtime.h>

// ---------------------------------------------------------------------------
// CDNA5 (gfx1250, wave32) fused cross-attention:
//   1) cvt kernels: f32 -> f16 (activations), f32 -> f16 transposed (weights)
//   2) WMMA GEMM (v_wmma_f32_16x16x32_f16), 32x32 tile per wave (2x2 subtiles)
//      for K/Q/V projections + output projection (f32 + bias epilogue)
//   3) flash-attention kernel: per-wave 16-query tile, key blocks of 32,
//      K block staged into LDS via the Tensor Data Mover (tensor_load_to_lds,
//      s_wait_tensorcnt), V staged transposed, online softmax, P@V via WMMA.
// ---------------------------------------------------------------------------

typedef __attribute__((ext_vector_type(16))) _Float16     v16h;
typedef __attribute__((ext_vector_type(8)))  _Float16     v8h;
typedef __attribute__((ext_vector_type(8)))  float        v8f;
typedef __attribute__((ext_vector_type(4)))  unsigned int v4u;
typedef __attribute__((ext_vector_type(8)))  int          v8i;
typedef __attribute__((ext_vector_type(4)))  int          v4i;

// Load one 16-bit WMMA fragment (A-layout or symmetric B-layout):
// lane holds 8 contiguous halves at p[0..7] (VGPRs 0-3) and p[16..23] (VGPRs 4-7).
__device__ __forceinline__ v16h load_frag16(const _Float16* p) {
  v8h lo = *(const v8h*)p;
  v8h hi = *(const v8h*)(p + 16);
  return __builtin_shufflevector(lo, hi,
      0, 1, 2, 3, 4, 5, 6, 7, 8, 9, 10, 11, 12, 13, 14, 15);
}

// ------------------------------ conversions --------------------------------

__global__ void __launch_bounds__(256) cvt_f32_f16(
    const float* __restrict__ in, _Float16* __restrict__ out, int n) {
  int i = blockIdx.x * blockDim.x + threadIdx.x;
  if (i < n) out[i] = (_Float16)in[i];
}

// in: K x N (row-major) -> out: N x K (row-major), i.e. transposed
__global__ void __launch_bounds__(256) cvt_transpose_f32_f16(
    const float* __restrict__ in, _Float16* __restrict__ out, int K, int N) {
  int i = blockIdx.x * blockDim.x + threadIdx.x;
  if (i < K * N) {
    int k = i / N;
    int n = i - k * N;
    out[(size_t)n * K + k] = (_Float16)in[i];
  }
}

// ------------------------------ WMMA GEMM ----------------------------------
// C[M,N] = A[M,K] * BT[N,K]^T   (both operands f16 row-major, f32 accumulate)
// One 32x32 tile per wave (2x2 WMMA subtiles -> each fragment feeds 2 WMMAs).
// 8 waves (256 threads, wave32) per block.
// If Cf32 != nullptr: write f32 (+bias). Else write f16 to C16.
__global__ void __launch_bounds__(256) wmma_gemm_kernel(
    const _Float16* __restrict__ A, const _Float16* __restrict__ BT,
    _Float16* __restrict__ C16, float* __restrict__ Cf32,
    const float* __restrict__ bias, int M, int N, int K) {
  const int wave = threadIdx.x >> 5;
  const int lane = threadIdx.x & 31;
  const int rowg = lane & 15;            // M (for A) / N (for B) within subtile
  const int khalf = lane >> 4;
  const int koff = khalf << 3;           // K offset 0 or 8 per lane half

  const int tile = blockIdx.x * 8 + wave;
  const int tiles_n = N >> 5;
  const int mtile = tile / tiles_n;
  const int ntile = tile - mtile * tiles_n;

  const _Float16* ap0 = A  + (size_t)(mtile * 32 + rowg) * K + koff;
  const _Float16* ap1 = ap0 + (size_t)16 * K;
  const _Float16* bp0 = BT + (size_t)(ntile * 32 + rowg) * K + koff;
  const _Float16* bp1 = bp0 + (size_t)16 * K;

  v8f acc00 = {}, acc01 = {}, acc10 = {}, acc11 = {};
  for (int kc = 0; kc < K; kc += 32) {
    __builtin_prefetch(ap0 + kc + 128, 0, 1);   // global_prefetch_b8
    __builtin_prefetch(bp0 + kc + 128, 0, 1);
    v16h a0 = load_frag16(ap0 + kc);
    v16h a1 = load_frag16(ap1 + kc);
    v16h b0 = load_frag16(bp0 + kc);
    v16h b1 = load_frag16(bp1 + kc);
    acc00 = __builtin_amdgcn_wmma_f32_16x16x32_f16(false, a0, false, b0, (short)0, acc00, false, false);
    acc01 = __builtin_amdgcn_wmma_f32_16x16x32_f16(false, a0, false, b1, (short)0, acc01, false, false);
    acc10 = __builtin_amdgcn_wmma_f32_16x16x32_f16(false, a1, false, b0, (short)0, acc10, false, false);
    acc11 = __builtin_amdgcn_wmma_f32_16x16x32_f16(false, a1, false, b1, (short)0, acc11, false, false);
  }

#pragma unroll
  for (int i = 0; i < 2; ++i) {
#pragma unroll
    for (int j = 0; j < 2; ++j) {
      const v8f& acc = (i == 0) ? (j == 0 ? acc00 : acc01)
                                : (j == 0 ? acc10 : acc11);
      const int col = ntile * 32 + j * 16 + rowg;
      const float bv = bias ? bias[col] : 0.f;
#pragma unroll
      for (int v = 0; v < 8; ++v) {
        int row = mtile * 32 + i * 16 + v + (khalf << 3);
        if (Cf32) Cf32[(size_t)row * N + col] = acc[v] + bv;
        else      C16 [(size_t)row * N + col] = (_Float16)acc[v];
      }
    }
  }
}

// --------------------------- TDM helper ------------------------------------
// Issue a 2D TDM load: tile (tile_d0 halves x tile_d1 rows) from a row-major
// f16 tensor with row stride (in halves) into LDS, padding the LDS destination
// by 8 DWORDs (16 halves) after every 128 DWORDs (256 halves) -> 272-half rows.
__device__ __forceinline__ void tdm_load_k_block(
    const _Float16* gaddr, unsigned lds_off) {
  unsigned long long ga = (unsigned long long)(const void*)gaddr;
  // group0: count=1 | lds_addr | global_addr | type=2
  v4u g0 = { 1u,
             lds_off,
             (unsigned)(ga & 0xffffffffu),
             (unsigned)((ga >> 32) & 0x01ffffffu) | 0x80000000u };
  // group1: data_size=2B (1<<16), pad_enable (1<<20),
  //         pad_interval=128 DWORDs (6<<22), pad_amount=8 DWORDs (7<<25)
  unsigned w0 = (1u << 16) | (1u << 20) | (6u << 22) | (7u << 25);
  unsigned tensor_d0 = 2048u;          // full row length (halves)
  unsigned tensor_d1 = 4096u;          // rows
  unsigned tile_d0 = 256u;             // halves per tile row
  unsigned tile_d1 = 32u;              // tile rows
  unsigned stride_d0 = 2048u;          // row stride (halves)
  v8i g1 = { (int)w0,
             (int)((tensor_d0 & 0xffffu) << 16),            // dim0[15:0] @ [63:48]
             (int)(((tensor_d0 >> 16) & 0xffffu) |          // dim0[31:16] @ [79:64]
                   ((tensor_d1 & 0xffffu) << 16)),          // dim1[15:0] @ [95:80]
             (int)(((tensor_d1 >> 16) & 0xffffu) |          // dim1[31:16] @ [111:96]
                   (tile_d0 << 16)),                        // tile_dim0 @ [127:112]
             (int)(tile_d1),                                // tile_dim1 @ [143:128]
             (int)(stride_d0),                              // dim0_stride[31:0] @ [191:160]
             0,                                             // dim0_stride[47:32], dim1_stride lo
             0 };
  v4i gz = { 0, 0, 0, 0 };
#if defined(__clang_major__) && (__clang_major__ >= 23)
  v8i gz8 = { 0, 0, 0, 0, 0, 0, 0, 0 };
  __builtin_amdgcn_tensor_load_to_lds(g0, g1, gz, gz, gz8, 0);
#else
  __builtin_amdgcn_tensor_load_to_lds(g0, g1, gz, gz, 0);
#endif
}

// --------------------------- flash attention -------------------------------
// Q,K,V layouts: (a*1024 + seq) x 2048, head h occupies columns [h*256, h*256+256).
// Grid: (a*8 + h, ytile); block = 256 threads = 8 waves; wave owns 16 queries.
// Key blocks of 32 so the P@V contraction matches WMMA K=32 exactly.
__global__ void __launch_bounds__(256) flash_attn_kernel(
    const _Float16* __restrict__ Q, const _Float16* __restrict__ Km,
    const _Float16* __restrict__ Vm, _Float16* __restrict__ Om) {
  // LDS strides padded to 16B multiples so b128 DS accesses are legal.
  // KLDS row stride 272 = 256 + 16 halves, exactly the TDM pad pattern.
  __shared__ __align__(16) _Float16 KLDS[32][272];     // key-major (32 x 256)
  __shared__ __align__(16) _Float16 VT[256][48];       // transposed: [c][key]
  __shared__ __align__(16) _Float16 PLDS[8][16][48];   // per-wave P (16 x 32)

  const int a = blockIdx.x >> 3;
  const int h = blockIdx.x & 7;
  const int wave = threadIdx.x >> 5;
  const int lane = threadIdx.x & 31;
  const int rowg = lane & 15;
  const int khalf = lane >> 4;
  const int koff = khalf << 3;
  const int q0 = blockIdx.y * 128 + wave * 16;

  const size_t brow0 = (size_t)a * 1024;
  const size_t hoff  = (size_t)h * 256;

  // Q fragments: 16 rows x 256 head-dim, 8 k-chunks of 32
  v16h qf[8];
  {
    const _Float16* qp = Q + (brow0 + q0 + rowg) * 2048 + hoff + koff;
#pragma unroll
    for (int kc = 0; kc < 8; ++kc) qf[kc] = load_frag16(qp + kc * 32);
  }

  v8f O[16] = {};              // 16 queries x 256 output accumulator
  float m_i[8], l_i[8];
#pragma unroll
  for (int v = 0; v < 8; ++v) { m_i[v] = -1e30f; l_i[v] = 0.f; }

  const float scale = 0.17677669529663687f;  // 1/sqrt(256/8)

  // cooperative-load coordinates for V: 32 key rows x 8 c-chunks of 32
  const int kr = threadIdx.x >> 3;
  const int cc = (threadIdx.x & 7) * 32;
  const unsigned klds_off = (unsigned)(size_t)(void*)&KLDS[0][0];

  for (int kb = 0; kb < 32; ++kb) {
    // ---- TDM: DMA the 32x256 K block into padded LDS rows (one issuer) ----
    if (threadIdx.x == 0) {
      tdm_load_k_block(Km + (brow0 + kb * 32) * 2048 + hoff, klds_off);
      __builtin_amdgcn_s_wait_tensorcnt(0);
    }
    // ---- stage V block transposed into LDS ----
    const _Float16* vp = Vm + (brow0 + kb * 32 + kr) * 2048 + hoff + cc;
#pragma unroll
    for (int i = 0; i < 4; ++i) {
      v8h t = *(const v8h*)(vp + 8 * i);
#pragma unroll
      for (int j = 0; j < 8; ++j) VT[cc + 8 * i + j][kr] = t[j];
    }
    __syncthreads();

    // ---- S = Q @ K^T : 16 queries x 32 keys as two 16x16 f32 tiles ----
    v8f s0 = {}, s1 = {};
#pragma unroll
    for (int kc = 0; kc < 8; ++kc) {
      v16h b0 = load_frag16(&KLDS[rowg][kc * 32 + koff]);
      v16h b1 = load_frag16(&KLDS[16 + rowg][kc * 32 + koff]);
      s0 = __builtin_amdgcn_wmma_f32_16x16x32_f16(
          false, qf[kc], false, b0, (short)0, s0, false, false);
      s1 = __builtin_amdgcn_wmma_f32_16x16x32_f16(
          false, qf[kc], false, b1, (short)0, s1, false, false);
    }

    // ---- online softmax (row = lane-half + element; cols across 16 lanes) ----
#pragma unroll
    for (int v = 0; v < 8; ++v) {
      float a0 = s0[v] * scale, a1 = s1[v] * scale;
      float mv = fmaxf(a0, a1);
#pragma unroll
      for (int off = 1; off < 16; off <<= 1) mv = fmaxf(mv, __shfl_xor(mv, off, 16));
      float mnew = fmaxf(m_i[v], mv);
      float alpha = __expf(m_i[v] - mnew);
      m_i[v] = mnew;
      float p0 = __expf(a0 - mnew);
      float p1 = __expf(a1 - mnew);
      float rs = p0 + p1;
#pragma unroll
      for (int off = 1; off < 16; off <<= 1) rs += __shfl_xor(rs, off, 16);
      l_i[v] = l_i[v] * alpha + rs;
#pragma unroll
      for (int t = 0; t < 16; ++t) O[t][v] *= alpha;
      // stage P into per-wave LDS (C-layout -> memory) for A-fragment reload
      PLDS[wave][v + khalf * 8][rowg]      = (_Float16)p0;
      PLDS[wave][v + khalf * 8][16 + rowg] = (_Float16)p1;
    }

    // ---- O += P(16x32) @ V(32x256), 16 WMMA over the c tiles ----
    v16h pf = load_frag16(&PLDS[wave][rowg][koff]);
#pragma unroll
    for (int n = 0; n < 16; ++n) {
      v16h vf = load_frag16(&VT[n * 16 + rowg][koff]);
      O[n] = __builtin_amdgcn_wmma_f32_16x16x32_f16(
          false, pf, false, vf, (short)0, O[n], false, false);
    }
    __syncthreads();
  }

  // ---- epilogue: O / l, store f16 into attn buffer (a*1024+q) x 2048 ----
#pragma unroll
  for (int v = 0; v < 8; ++v) {
    float inv = 1.f / l_i[v];
    int row = q0 + v + khalf * 8;
    _Float16* op = Om + (brow0 + row) * 2048 + hoff + rowg;
#pragma unroll
    for (int n = 0; n < 16; ++n) op[n * 16] = (_Float16)(O[n][v] * inv);
  }
}

// ------------------------------- launcher ----------------------------------

extern "C" void kernel_launch(void* const* d_in, const int* in_sizes, int n_in,
                              void* d_out, int out_size, void* d_ws, size_t ws_size,
                              hipStream_t stream) {
  const float* batch  = (const float*)d_in[0];  // (4,1024,256)
  const float* latent = (const float*)d_in[1];  // (4,1024,256)
  const float* Wk     = (const float*)d_in[2];  // (256,2048)
  const float* Wq     = (const float*)d_in[3];
  const float* Wv     = (const float*)d_in[4];
  const float* Wu     = (const float*)d_in[5];  // (2048,256)
  const float* bu     = (const float*)d_in[6];  // (256,)
  float* out = (float*)d_out;                   // (4,1024,256) f32

  _Float16* w = (_Float16*)d_ws;
  size_t o = 0;
  _Float16* batch16  = w + o; o += (size_t)4096 * 256;
  _Float16* latent16 = w + o; o += (size_t)4096 * 256;
  _Float16* WkT = w + o; o += (size_t)2048 * 256;   // (2048 x 256) = Wk^T
  _Float16* WqT = w + o; o += (size_t)2048 * 256;
  _Float16* WvT = w + o; o += (size_t)2048 * 256;
  _Float16* WuT = w + o; o += (size_t)256 * 2048;   // (256 x 2048) = Wu^T
  _Float16* K16 = w + o; o += (size_t)4096 * 2048;
  _Float16* Q16 = w + o; o += (size_t)4096 * 2048;
  _Float16* V16 = w + o; o += (size_t)4096 * 2048;
  _Float16* A16 = w + o; o += (size_t)4096 * 2048;  // attention output

  // 1) conversions
  cvt_f32_f16<<<4096, 256, 0, stream>>>(batch,  batch16,  4096 * 256);
  cvt_f32_f16<<<4096, 256, 0, stream>>>(latent, latent16, 4096 * 256);
  cvt_transpose_f32_f16<<<2048, 256, 0, stream>>>(Wk, WkT, 256, 2048);
  cvt_transpose_f32_f16<<<2048, 256, 0, stream>>>(Wq, WqT, 256, 2048);
  cvt_transpose_f32_f16<<<2048, 256, 0, stream>>>(Wv, WvT, 256, 2048);
  cvt_transpose_f32_f16<<<2048, 256, 0, stream>>>(Wu, WuT, 2048, 256);

  // 2) projections: (4096x256)@(256x2048) -> f16
  //    32x32 wave tiles: (4096/32)*(2048/32) = 8192 tiles -> 1024 blocks
  wmma_gemm_kernel<<<1024, 256, 0, stream>>>(batch16,  WkT, K16, nullptr, nullptr, 4096, 2048, 256);
  wmma_gemm_kernel<<<1024, 256, 0, stream>>>(latent16, WqT, Q16, nullptr, nullptr, 4096, 2048, 256);
  wmma_gemm_kernel<<<1024, 256, 0, stream>>>(batch16,  WvT, V16, nullptr, nullptr, 4096, 2048, 256);

  // 3) fused flash attention: grid (a*h = 32, ytiles = 1024/128 = 8)
  flash_attn_kernel<<<dim3(32, 8), 256, 0, stream>>>(Q16, K16, V16, A16);

  // 4) output projection: (4096x2048)@(2048x256) + bias -> f32
  //    (4096/32)*(256/32) = 1024 tiles -> 128 blocks
  wmma_gemm_kernel<<<128, 256, 0, stream>>>(A16, WuT, nullptr, out, bu, 4096, 256, 2048);
}
